// EncoderBlock_14508399526301
// MI455X (gfx1250) — matrix operand order repaired and checked by reading, not compiled, not run on hardware
//
#include <hip/hip_runtime.h>
#include <cstdint>
#include <cstddef>

// ---------------- types / helpers ----------------
typedef __bf16 bf16;
typedef bf16  v16bf __attribute__((ext_vector_type(16)));
typedef float v8f   __attribute__((ext_vector_type(8)));

#define DEV __device__ __forceinline__

// sizes
static constexpr int EMBED = 512, HIDDEN = 512, VOCAB = 30000;
static constexpr int NB = 16, NS = 128, NK = 49, NT5 = 5;
static constexpr int ROWS = NB * NS;          // 2048
static constexpr int KT   = HIDDEN / 32;      // 16 k-tiles of 32
static constexpr int MT   = ROWS / 16;        // 128 m-tiles

DEV unsigned short f2bf(float f) {
    unsigned u = __float_as_uint(f);
    unsigned r = u + 0x7fffu + ((u >> 16) & 1u);     // round-to-nearest-even
    if ((u & 0x7f800000u) == 0x7f800000u) r = u;     // inf/NaN passthrough
    return (unsigned short)(r >> 16);
}

DEV float wsum(float v) {
#pragma unroll
    for (int o = 16; o > 0; o >>= 1) v += __shfl_xor(v, o, 32);
    return v;
}
DEV float wmaxr(float v) {
#pragma unroll
    for (int o = 16; o > 0; o >>= 1) v = fmaxf(v, __shfl_xor(v, o, 32));
    return v;
}

DEV v8f wmma_bf16(v16bf a, v16bf b, v8f c) {
    return __builtin_amdgcn_wmma_f32_16x16x32_bf16(false, a, false, b, (short)0, c,
                                                   false, false);
}

// A fragment: 16x32 bf16 tile from row-major [M,K] bf16 matrix.
// ISA layout: lanes 0-15 -> rows m0..m0+15, K {0..7,16..23}; lanes 16-31 same rows, K {8..15,24..31}.
DEV v16bf load_a(const unsigned short* A, int ldk, int mt, int kt, int lane) {
    int row = mt * 16 + (lane & 15);
    int k0  = kt * 32 + ((lane & 16) ? 8 : 0);
    const unsigned short* p = A + (size_t)row * ldk + k0;
    union { v16bf v; uint4 u[2]; } f;
    f.u[0] = *(const uint4*)(p);
    f.u[1] = *(const uint4*)(p + 16);
    return f.v;
}

// B fragment: pre-packed tiles; lane l reads 16 contiguous bf16 = (n = n0+(l&15), k = k(l>>4,p) for p = 0..15).
DEV v16bf load_b(const unsigned short* Bpk, int tile, int lane) {
    const unsigned short* p = Bpk + (size_t)tile * 512 + lane * 16;
    union { v16bf v; uint4 u[2]; } f;
    f.u[0] = *(const uint4*)(p);
    f.u[1] = *(const uint4*)(p + 8);
    return f.v;
}

// ---------------- weight packing: W[N,K] fp32 row-major -> bf16 B-fragment tiles ----------------
// dst[(ntile*nTK + kt)*512 + l*16 + p] = bf16(W[(ntile*16+(l&15))][kt*32+k(l>>4,p)]), k(h,p) = p<8 ? 8h+p : 16+8h+p-8
__global__ __launch_bounds__(256) void pack_w_kernel(const float* __restrict__ src,
                                                     unsigned short* __restrict__ dst,
                                                     int Nrows, int Kdim) {
    int idx = blockIdx.x * 256 + threadIdx.x;
    int total = Nrows * Kdim;
    if (idx >= total) return;
    int tile = idx >> 9;
    int tloc = idx & 511;
    int l  = tloc >> 4;
    int p  = tloc & 15;
    int nTK = Kdim >> 5;
    int kt = tile % nTK;
    int ntile = tile / nTK;
    int hh = l >> 4;
    int kk = (p < 8) ? (8 * hh + p) : (16 + 8 * hh + (p - 8));
    float v = src[(size_t)(ntile * 16 + (l & 15)) * Kdim + kt * 32 + kk];
    dst[idx] = f2bf(v);
}

// ---------------- ZV / QT precompute (per batch, shared by all S) ----------------
__global__ __launch_bounds__(256) void zvqt_kernel(const float* __restrict__ V,
                                                   const float* __restrict__ T,
                                                   const float* __restrict__ W_ZV,
                                                   const float* __restrict__ b_ZV,
                                                   const float* __restrict__ W_QT,
                                                   const float* __restrict__ b_QT,
                                                   float* __restrict__ ZVws,
                                                   float* __restrict__ QTws) {
    int w = threadIdx.x >> 5, lane = threadIdx.x & 31;
    int row = blockIdx.x * 8 + w;          // 0..863
    if (row < NB * NK) {
        int b = row / NK, k = row % NK;
        float vl[16];
#pragma unroll
        for (int i = 0; i < 16; ++i) vl[i] = V[((size_t)(b * NK + k)) * HIDDEN + lane + 32 * i];
        for (int j = 0; j < NK; ++j) {
            const float* wr = W_ZV + (size_t)j * HIDDEN;
            float acc = 0.f;
#pragma unroll
            for (int i = 0; i < 16; ++i) acc += vl[i] * wr[lane + 32 * i];
            acc = wsum(acc);
            if (lane == 0) ZVws[(size_t)(b * NK + k) * NK + j] = acc + b_ZV[j];
        }
    } else {
        int r2 = row - NB * NK;
        int b = r2 / NT5, t = r2 % NT5;
        float tl[16];
#pragma unroll
        for (int i = 0; i < 16; ++i) tl[i] = T[((size_t)(b * NT5 + t)) * EMBED + lane + 32 * i];
        for (int u = 0; u < NT5; ++u) {
            const float* wr = W_QT + (size_t)u * EMBED;
            float acc = 0.f;
#pragma unroll
            for (int i = 0; i < 16; ++i) acc += tl[i] * wr[lane + 32 * i];
            acc = wsum(acc);
            if (lane == 0) QTws[(size_t)(b * NT5 + t) * NT5 + u] = acc + b_QT[u];
        }
    }
}

// ---------------- attention: per-row z_t, q_t, plus bf16 h; qh0 for batch 0 ----------------
__global__ __launch_bounds__(256) void attend_kernel(
    const float* __restrict__ h_t, const float* __restrict__ V, const float* __restrict__ T,
    const float* __restrict__ ZVws, const float* __restrict__ QTws,
    const float* __restrict__ W_Zh, const float* __restrict__ b_Zh,
    const float* __restrict__ W_az, const float* __restrict__ b_az,
    const float* __restrict__ W_Qh, const float* __restrict__ b_Qh,
    const float* __restrict__ W_bq, const float* __restrict__ b_bq,
    unsigned short* __restrict__ zbf, unsigned short* __restrict__ qbf,
    unsigned short* __restrict__ hbf, float* __restrict__ qh0ws) {
    __shared__ float smZh[8][NK];
    __shared__ float smAl[8][NK];
    int w = threadIdx.x >> 5, lane = threadIdx.x & 31;
    int row = blockIdx.x * 8 + w;          // 0..2047
    int b = row >> 7, s = row & 127;

    float hl[16];
#pragma unroll
    for (int i = 0; i < 16; ++i) hl[i] = h_t[(size_t)row * HIDDEN + lane + 32 * i];

    // Zh[j] = h . W_Zh[j] + b_Zh[j]
    for (int j = 0; j < NK; ++j) {
        const float* wr = W_Zh + (size_t)j * HIDDEN;
        float acc = 0.f;
#pragma unroll
        for (int i = 0; i < 16; ++i) acc += hl[i] * wr[lane + 32 * i];
        acc = wsum(acc);
        if (lane == 0) smZh[w][j] = acc + b_Zh[j];
    }
    // Qh[u]
    float qh[NT5];
#pragma unroll
    for (int u = 0; u < NT5; ++u) {
        const float* wr = W_Qh + (size_t)u * HIDDEN;
        float acc = 0.f;
#pragma unroll
        for (int i = 0; i < 16; ++i) acc += hl[i] * wr[lane + 32 * i];
        qh[u] = wsum(acc) + b_Qh[u];
    }
    __syncthreads();

    // visual scores over K=49 (lane covers k and k+32)
    float az = b_az[0];
    int k1 = lane;
    float v1 = az;
    {
        const float* zb = ZVws + (size_t)(b * NK + k1) * NK;
        for (int j = 0; j < NK; ++j) v1 += tanhf(zb[j] + smZh[w][j]) * W_az[j];
    }
    int k2 = lane + 32;
    bool ok2 = (k2 < NK);
    float v2 = -INFINITY;
    if (ok2) {
        v2 = az;
        const float* zb = ZVws + (size_t)(b * NK + k2) * NK;
        for (int j = 0; j < NK; ++j) v2 += tanhf(zb[j] + smZh[w][j]) * W_az[j];
    }
    float m  = wmaxr(fmaxf(v1, v2));
    float e1 = __expf(v1 - m);
    float e2 = ok2 ? __expf(v2 - m) : 0.f;
    float den = wsum(e1 + e2);
    smAl[w][k1] = e1 / den;
    if (ok2) smAl[w][k2] = e2 / den;
    __syncthreads();

    // topic softmax (redundant across lanes; all small)
    float bq = b_bq[0];
    float tv[NT5];
#pragma unroll
    for (int t = 0; t < NT5; ++t) {
        float a = bq;
#pragma unroll
        for (int u = 0; u < NT5; ++u)
            a += tanhf(QTws[(size_t)(b * NT5 + t) * NT5 + u] + qh[u]) * W_bq[u];
        tv[t] = a;
    }
    float tm = tv[0];
#pragma unroll
    for (int t = 1; t < NT5; ++t) tm = fmaxf(tm, tv[t]);
    float beta[NT5], tsum = 0.f;
#pragma unroll
    for (int t = 0; t < NT5; ++t) { beta[t] = __expf(tv[t] - tm); tsum += beta[t]; }
#pragma unroll
    for (int t = 0; t < NT5; ++t) beta[t] /= tsum;

    // z_t, q_t, h -> bf16
#pragma unroll 4
    for (int i = 0; i < 16; ++i) {
        int e = lane + 32 * i;
        float zacc = 0.f;
        for (int k = 0; k < NK; ++k)
            zacc += smAl[w][k] * V[(size_t)(b * NK + k) * HIDDEN + e];
        float qacc = 0.f;
#pragma unroll
        for (int t = 0; t < NT5; ++t)
            qacc += beta[t] * T[(size_t)(b * NT5 + t) * EMBED + e];
        size_t o = (size_t)row * HIDDEN + e;
        zbf[o] = f2bf(zacc);
        qbf[o] = f2bf(qacc);
        hbf[o] = f2bf(hl[i]);
    }
    if (b == 0 && lane == 0) {
#pragma unroll
        for (int u = 0; u < NT5; ++u) qh0ws[s * NT5 + u] = qh[u];
    }
}

// ---------------- register-pipelined WMMA pass (small GEMMs) ----------------
template <int NTILE, int NTK>
DEV void gemm_pass(const unsigned short* __restrict__ A,
                   const unsigned short* __restrict__ Bpk,
                   v8f* acc, int mt, int nt0, int lane) {
    constexpr int ldk = NTK * 32;
    v16bf a0 = load_a(A, ldk, mt, 0, lane);
    v16bf b0[NTILE];
#pragma unroll
    for (int t = 0; t < NTILE; ++t) b0[t] = load_b(Bpk, (nt0 + t) * NTK, lane);

#pragma unroll
    for (int kt = 0; kt < NTK - 1; ++kt) {
        v16bf a1 = load_a(A, ldk, mt, kt + 1, lane);
        v16bf b1[NTILE];
#pragma unroll
        for (int t = 0; t < NTILE; ++t) b1[t] = load_b(Bpk, (nt0 + t) * NTK + kt + 1, lane);
#pragma unroll
        for (int t = 0; t < NTILE; ++t) acc[t] = wmma_bf16(a0, b0[t], acc[t]);
        a0 = a1;
#pragma unroll
        for (int t = 0; t < NTILE; ++t) b0[t] = b1[t];
    }
#pragma unroll
    for (int t = 0; t < NTILE; ++t) acc[t] = wmma_bf16(a0, b0[t], acc[t]);
}

template <int NTILE, bool TANH, bool DUAL>
DEV void gemm_core(const unsigned short* A0, const unsigned short* B0,
                   const unsigned short* A1, const unsigned short* B1,
                   const float* bias0, const float* bias1,
                   float* out, int N, int mt, int nt0, int lane) {
    v8f acc[NTILE] = {};
    gemm_pass<NTILE, KT>(A0, B0, acc, mt, nt0, lane);
    if (DUAL) gemm_pass<NTILE, KT>(A1, B1, acc, mt, nt0, lane);

    int n  = lane & 15;
    int mb = (lane >> 4) * 8;
#pragma unroll
    for (int t = 0; t < NTILE; ++t) {
        int n0 = (nt0 + t) * 16;
        float bv = bias0[n0 + n];
        if (DUAL) bv += bias1[n0 + n];
#pragma unroll
        for (int r = 0; r < 8; ++r) {
            float v = acc[t][r] + bv;
            if (TANH) v = tanhf(v);
            out[(size_t)(mt * 16 + mb + r) * N + n0 + n] = v;
        }
    }
}

// r_t = tanh(z @ Wsz^T + b_sz)
__global__ __launch_bounds__(256) void gemm_r_kernel(const unsigned short* zbf,
                                                     const unsigned short* Wsz_pk,
                                                     const float* b_sz, float* rt) {
    int lane = threadIdx.x & 31;
    int wid = blockIdx.x * 8 + (threadIdx.x >> 5);     // 0..1023
    int mt = wid >> 3, nt0 = (wid & 7) * 4;
    gemm_core<4, true, false>(zbf, Wsz_pk, nullptr, nullptr, b_sz, nullptr,
                              rt, HIDDEN, mt, nt0, lane);
}

// s_t = tanh(q @ Wsq^T + h @ Wsh^T + b_sq + b_sh)
__global__ __launch_bounds__(256) void gemm_s_kernel(const unsigned short* qbf,
                                                     const unsigned short* Wsq_pk,
                                                     const unsigned short* hbf,
                                                     const unsigned short* Wsh_pk,
                                                     const float* b_sq, const float* b_sh,
                                                     float* st) {
    int lane = threadIdx.x & 31;
    int wid = blockIdx.x * 8 + (threadIdx.x >> 5);
    int mt = wid >> 3, nt0 = (wid & 7) * 4;
    gemm_core<4, true, true>(qbf, Wsq_pk, hbf, Wsh_pk, b_sq, b_sh,
                             st, EMBED, mt, nt0, lane);
}

// ---------------- big projection with async LDS staging of B tiles ----------------
// Block = 8 waves over 8 consecutive m-tiles sharing one 5-tile n-group.
// B tiles for k-step kt+1 stream into LDS via global_load_async_to_lds_b128
// (ASYNCcnt) while all 8 waves run WMMAs on k-step kt from the other buffer.
__global__ __launch_bounds__(256) void gemm_mlp_kernel(const unsigned short* __restrict__ cbf,
                                                       const unsigned short* __restrict__ Wmlp_pk,
                                                       const float* __restrict__ b_mlp,
                                                       float* __restrict__ out) {
    __shared__ unsigned short smB[2][NT5 * 512];       // 2 x 5120 B
    const int tid  = threadIdx.x;
    const int w    = tid >> 5, lane = tid & 31;
    const int mt   = (blockIdx.x & 15) * 8 + w;        // 128 m-tiles / 8 per block
    const int nt0  = (blockIdx.x >> 4) * NT5;          // 375 n-groups

    // async-copy one k-step of 5 B tiles (5120B = 320 x 16B chunks) into smB[stage]
    auto issue_stage = [&](int kt, int stage) {
        {
            int c = tid;                               // chunks 0..255
            int t = c >> 6, o = c & 63;
            const unsigned short* gp =
                Wmlp_pk + ((size_t)((nt0 + t) * KT + kt) << 9) + (o << 3);
            unsigned lda = (unsigned)(unsigned long long)(const void*)&smB[stage][(size_t)c << 3];
            asm volatile("global_load_async_to_lds_b128 %0, %1, off"
                         :: "v"(lda), "v"(gp) : "memory");
        }
        if (tid < 64) {                                // chunks 256..319 (waves 0,1 only)
            int c = 256 + tid;
            int t = c >> 6, o = c & 63;
            const unsigned short* gp =
                Wmlp_pk + ((size_t)((nt0 + t) * KT + kt) << 9) + (o << 3);
            unsigned lda = (unsigned)(unsigned long long)(const void*)&smB[stage][(size_t)c << 3];
            asm volatile("global_load_async_to_lds_b128 %0, %1, off"
                         :: "v"(lda), "v"(gp) : "memory");
        }
    };

    v8f acc[NT5] = {};
    issue_stage(0, 0);
    v16bf a_cur = load_a(cbf, 512, mt, 0, lane);

#pragma unroll
    for (int kt = 0; kt < KT; ++kt) {
        const int stage = kt & 1;
        v16bf a_next;
        if (kt + 1 < KT) {
            issue_stage(kt + 1, stage ^ 1);            // refill other buffer
            a_next = load_a(cbf, 512, mt, kt + 1, lane);
            // wait for *previous* stage copies (ours), leave the new stage in flight
            if (w < 2) asm volatile("s_wait_asynccnt 0x2");
            else       asm volatile("s_wait_asynccnt 0x1");
        } else {
            asm volatile("s_wait_asynccnt 0x0");
        }
        __syncthreads();                                // publish stage data block-wide

#pragma unroll
        for (int t = 0; t < NT5; ++t) {
            union { v16bf v; uint4 u[2]; } f;
            const unsigned short* p = &smB[stage][t * 512 + lane * 16];
            f.u[0] = *(const uint4*)(p);
            f.u[1] = *(const uint4*)(p + 8);
            acc[t] = wmma_bf16(a_cur, f.v, acc[t]);
        }
        __syncthreads();                                // reads done before buffer refill
        if (kt + 1 < KT) a_cur = a_next;
    }

    // writeout with bias; C/D layout: n = lane&15, M = r + 8*(lane>>4)
    int n  = lane & 15;
    int mb = (lane >> 4) * 8;
#pragma unroll
    for (int t = 0; t < NT5; ++t) {
        int n0 = (nt0 + t) * 16;
        float bv = b_mlp[n0 + n];
#pragma unroll
        for (int r = 0; r < 8; ++r)
            out[(size_t)(mt * 16 + mb + r) * VOCAB + n0 + n] = acc[t][r] + bv;
    }
}

// ---------------- sentinel gate gamma[s] (batch 0 diagonal) ----------------
__global__ __launch_bounds__(256) void sentinel_kernel(
    const float* __restrict__ st, const float* __restrict__ rt,
    const float* __restrict__ qh0ws,
    const float* __restrict__ W_Ss, const float* __restrict__ b_Ss,
    const float* __restrict__ W_Sr, const float* __restrict__ b_Sr,
    const float* __restrict__ W_bq, const float* __restrict__ b_bq,
    const int* __restrict__ epoch, float* __restrict__ gama) {
    int w = threadIdx.x >> 5, lane = threadIdx.x & 31;
    int s = blockIdx.x * 8 + w;            // 0..127 (batch 0 rows == rows 0..127)
    float sl[16], rl[16];
#pragma unroll
    for (int i = 0; i < 16; ++i) {
        sl[i] = st[(size_t)s * EMBED + lane + 32 * i];
        rl[i] = rt[(size_t)s * EMBED + lane + 32 * i];
    }
    float scs = 0.f, scr = 0.f;
#pragma unroll
    for (int u = 0; u < NT5; ++u) {
        const float* ws = W_Ss + (size_t)u * EMBED;
        const float* wr = W_Sr + (size_t)u * EMBED;
        float d1 = 0.f, d2 = 0.f;
#pragma unroll
        for (int i = 0; i < 16; ++i) {
            d1 += sl[i] * ws[lane + 32 * i];
            d2 += rl[i] * wr[lane + 32 * i];
        }
        d1 = wsum(d1) + b_Ss[u] + qh0ws[s * NT5 + u];
        d2 = wsum(d2) + b_Sr[u] + qh0ws[s * NT5 + u];
        scs += tanhf(d1) * W_bq[u];
        scr += tanhf(d2) * W_bq[u];
    }
    float score_s = scs + b_bq[0];
    float score_r = scr + b_bq[0];
    float g = (*epoch <= 20) ? 1.0f : 1.f / (1.f + __expf(-(score_s - score_r)));
    if (lane == 0) gama[s] = g;
}

// ---------------- c_t = g*s + (1-g)*r -> bf16 ----------------
__global__ __launch_bounds__(256) void combine_kernel(const float* __restrict__ st,
                                                      const float* __restrict__ rt,
                                                      const float* __restrict__ gama,
                                                      unsigned short* __restrict__ cbf) {
    size_t idx = (size_t)blockIdx.x * 256 + threadIdx.x;    // < 2048*512
    int row = (int)(idx >> 9);
    int s = row & 127;
    float g = gama[s];
    float c = g * st[idx] + (1.f - g) * rt[idx];
    cbf[idx] = f2bf(c);
}

// ---------------- host-side launch ----------------
extern "C" void kernel_launch(void* const* d_in, const int* in_sizes, int n_in,
                              void* d_out, int out_size, void* d_ws, size_t ws_size,
                              hipStream_t stream) {
    (void)in_sizes; (void)n_in; (void)out_size; (void)ws_size;
    const int*   epoch = (const int*)d_in[0];
    const float* h_t   = (const float*)d_in[1];
    const float* V     = (const float*)d_in[2];
    const float* T     = (const float*)d_in[3];
    const float* W_ZV  = (const float*)d_in[4];
    const float* b_ZV  = (const float*)d_in[5];
    const float* W_Zh  = (const float*)d_in[6];
    const float* b_Zh  = (const float*)d_in[7];
    const float* W_az  = (const float*)d_in[8];
    const float* b_az  = (const float*)d_in[9];
    const float* W_QT  = (const float*)d_in[10];
    const float* b_QT  = (const float*)d_in[11];
    const float* W_Qh  = (const float*)d_in[12];
    const float* b_Qh  = (const float*)d_in[13];
    const float* W_bq  = (const float*)d_in[14];
    const float* b_bq  = (const float*)d_in[15];
    const float* W_sq  = (const float*)d_in[16];
    const float* b_sq  = (const float*)d_in[17];
    const float* W_sh  = (const float*)d_in[18];
    const float* b_sh  = (const float*)d_in[19];
    const float* W_Ss  = (const float*)d_in[20];
    const float* b_Ss  = (const float*)d_in[21];
    const float* W_Sr  = (const float*)d_in[22];
    const float* b_Sr  = (const float*)d_in[23];
    const float* W_sz  = (const float*)d_in[24];
    const float* b_sz  = (const float*)d_in[25];
    const float* W_mlp = (const float*)d_in[26];
    const float* b_mlp = (const float*)d_in[27];
    float* out = (float*)d_out;

    // workspace carve-out (256B aligned)
    char* ws = (char*)d_ws;
    size_t off = 0;
    auto take = [&](size_t bytes) -> char* {
        char* p = ws + off;
        off = (off + bytes + 255) & ~(size_t)255;
        return p;
    };
    float* ZVws  = (float*)take((size_t)NB * NK * NK * 4);
    float* QTws  = (float*)take((size_t)NB * NT5 * NT5 * 4);
    float* qh0ws = (float*)take((size_t)NS * NT5 * 4);
    float* gama  = (float*)take((size_t)NS * 4);
    unsigned short* zbf = (unsigned short*)take((size_t)ROWS * HIDDEN * 2);
    unsigned short* qbf = (unsigned short*)take((size_t)ROWS * HIDDEN * 2);
    unsigned short* hbf = (unsigned short*)take((size_t)ROWS * HIDDEN * 2);
    unsigned short* cbf = (unsigned short*)take((size_t)ROWS * EMBED * 2);
    float* rt = (float*)take((size_t)ROWS * EMBED * 4);
    float* st = (float*)take((size_t)ROWS * EMBED * 4);
    unsigned short* Wsz_pk  = (unsigned short*)take((size_t)EMBED * HIDDEN * 2);
    unsigned short* Wsq_pk  = (unsigned short*)take((size_t)EMBED * EMBED * 2);
    unsigned short* Wsh_pk  = (unsigned short*)take((size_t)EMBED * HIDDEN * 2);
    unsigned short* Wmlp_pk = (unsigned short*)take((size_t)VOCAB * EMBED * 2);

    // 1) pack weights to bf16 B-fragment tiles
    {
        int n = EMBED * HIDDEN;
        pack_w_kernel<<<(n + 255) / 256, 256, 0, stream>>>(W_sz, Wsz_pk, EMBED, HIDDEN);
        pack_w_kernel<<<(n + 255) / 256, 256, 0, stream>>>(W_sq, Wsq_pk, EMBED, EMBED);
        pack_w_kernel<<<(n + 255) / 256, 256, 0, stream>>>(W_sh, Wsh_pk, EMBED, HIDDEN);
        int nm = VOCAB * EMBED;
        pack_w_kernel<<<(nm + 255) / 256, 256, 0, stream>>>(W_mlp, Wmlp_pk, VOCAB, EMBED);
    }
    // 2) per-batch ZV/QT
    zvqt_kernel<<<(NB * (NK + NT5)) / 8, 256, 0, stream>>>(V, T, W_ZV, b_ZV, W_QT, b_QT,
                                                           ZVws, QTws);
    // 3) attention -> z,q,h bf16 + qh0
    attend_kernel<<<ROWS / 8, 256, 0, stream>>>(h_t, V, T, ZVws, QTws,
                                                W_Zh, b_Zh, W_az, b_az,
                                                W_Qh, b_Qh, W_bq, b_bq,
                                                zbf, qbf, hbf, qh0ws);
    // 4) r_t, s_t via bf16 WMMA GEMMs
    gemm_r_kernel<<<(MT * 8) / 8, 256, 0, stream>>>(zbf, Wsz_pk, b_sz, rt);
    gemm_s_kernel<<<(MT * 8) / 8, 256, 0, stream>>>(qbf, Wsq_pk, hbf, Wsh_pk, b_sq, b_sh, st);
    // 5) sentinel gate
    sentinel_kernel<<<NS / 8, 256, 0, stream>>>(st, rt, qh0ws, W_Ss, b_Ss, W_Sr, b_Sr,
                                                W_bq, b_bq, epoch, gama);
    // 6) combine -> c bf16
    combine_kernel<<<(ROWS * EMBED) / 256, 256, 0, stream>>>(st, rt, gama, cbf);
    // 7) big projection: (128/8) m-blocks x 375 n-groups = 6000 blocks
    gemm_mlp_kernel<<<(MT / 8) * (VOCAB / 80), 256, 0, stream>>>(cbf, Wmlp_pk, b_mlp, out);
}
